// MambaModel_62423054680256
// MI455X (gfx1250) — matrix operand order repaired and checked by reading, not compiled, not run on hardware
//
#include <hip/hip_runtime.h>
#include <hip/hip_bf16.h>

// ---------------- problem constants (match reference) ----------------
#define D_MODEL   1024
#define D_STATE   16
#define D_CONV    4
#define NUM_LAYERS 2
#define SEQ       1024
#define BATCH     2
#define D_INNER   2048          // EXPAND * D_MODEL
#define DT_RANK   64            // ceil(d_model/16)
#define BL        (BATCH*SEQ)   // 2048 rows for all GEMMs
#define NPAD_X    128           // dt_rank + 2*d_state = 96 padded to 128
#define NCHUNK    16            // scan chunks (parallel)
#define CHUNK_L   (SEQ / NCHUNK) // 64 sequential steps per chunk
#define LOG2E     1.4426950408889634f

// ---------------- CDNA5 WMMA types ----------------
typedef __attribute__((ext_vector_type(16))) __bf16 v16bf;
typedef __attribute__((ext_vector_type(8)))  float  v8f;

union FragAB { v16bf v; uint4 q[2]; };

__device__ __forceinline__ unsigned short f2bf(float f) {
    unsigned int u = __float_as_uint(f);
    unsigned int r = u + 0x7fffu + ((u >> 16) & 1u);   // round-to-nearest-even
    return (unsigned short)(r >> 16);
}
__device__ __forceinline__ float sigmoid_f(float x) { return 1.0f / (1.0f + __expf(-x)); }
__device__ __forceinline__ float silu_f(float x)    { return x * sigmoid_f(x); }
__device__ __forceinline__ float softplus_f(float x){ return (x > 20.0f) ? x : log1pf(__expf(x)); }

// =====================================================================
// Generic bf16 WMMA GEMM:  C[M,N] (f32) = A[M,K] (bf16, lda) * W[N,K]^T (bf16, ldb)
// Each wave computes a 64x64 tile via 4x4 v_wmma_f32_16x16x32_bf16 fragments.
// M,N multiples of 64; K multiple of 32; lda/ldb multiples of 8 (16B alignment).
// =====================================================================
__global__ __launch_bounds__(256) void gemm_bf16_nt(
    const unsigned short* __restrict__ A, int lda,
    const unsigned short* __restrict__ Bw, int ldb,
    float* __restrict__ C, int ldc,
    int M, int N, int K)
{
    const int wave = blockIdx.x * (blockDim.x >> 5) + (threadIdx.x >> 5);
    const int TN = N >> 6;
    const int TM = M >> 6;
    if (wave >= TM * TN) return;                 // uniform per wave (EXEC stays all-1)
    const int tm = wave / TN;
    const int tn = wave % TN;
    const int m0 = tm << 6;
    const int n0 = tn << 6;
    const int lane = threadIdx.x & 31;
    const int half = lane >> 4;                  // 0: lanes 0-15, 1: lanes 16-31
    const int l16  = lane & 15;

    v8f acc[4][4] = {};

    for (int k0 = 0; k0 < K; k0 += 32) {
        FragAB a[4], b[4];
        // A fragment 16x32: lane half 0 -> K {0..7,16..23}; half 1 -> K {8..15,24..31}
        #pragma unroll
        for (int mi = 0; mi < 4; ++mi) {
            const unsigned short* p =
                A + (size_t)(m0 + mi * 16 + l16) * lda + k0 + half * 8;
            a[mi].q[0] = *(const uint4*)(p);        // 8 bf16
            a[mi].q[1] = *(const uint4*)(p + 16);   // 8 bf16, +32B
        }
        // B fragment 32x16 (= W row): half 0 -> K {0..7,16..23}; half 1 -> K {8..15,24..31}
        #pragma unroll
        for (int ni = 0; ni < 4; ++ni) {
            const unsigned short* p =
                Bw + (size_t)(n0 + ni * 16 + l16) * ldb + k0 + half * 8;
            b[ni].q[0] = *(const uint4*)(p);
            b[ni].q[1] = *(const uint4*)(p + 16);
        }
        if (k0 + 32 < K) {   // prefetch next K-slab (global_prefetch_b8)
            __builtin_prefetch(A  + (size_t)(m0 + l16) * lda + k0 + 32, 0, 1);
            __builtin_prefetch(Bw + (size_t)(n0 + l16) * ldb + k0 + 32, 0, 1);
        }
        #pragma unroll
        for (int mi = 0; mi < 4; ++mi)
            #pragma unroll
            for (int ni = 0; ni < 4; ++ni)
                acc[mi][ni] = __builtin_amdgcn_wmma_f32_16x16x32_bf16(
                    false, a[mi].v, false, b[ni].v,
                    (short)0, acc[mi][ni], false, false);
    }

    // C/D layout: VGPR r -> row r (lanes 0-15) / row r+8 (lanes 16-31); lane&15 -> col
    #pragma unroll
    for (int mi = 0; mi < 4; ++mi)
        #pragma unroll
        for (int ni = 0; ni < 4; ++ni) {
            const int col = n0 + ni * 16 + l16;
            #pragma unroll
            for (int r = 0; r < 8; ++r) {
                const int row = m0 + mi * 16 + r + half * 8;
                C[(size_t)row * ldc + col] = acc[mi][ni][r];
            }
        }
}

// =====================================================================
// Elementwise / conversion kernels
// =====================================================================
__global__ void prep_x_kernel(const float* __restrict__ pe,
                              const float* __restrict__ cond,
                              float* __restrict__ x, unsigned short* __restrict__ xb)
{
    int i = blockIdx.x * blockDim.x + threadIdx.x;
    if (i >= BL * D_MODEL) return;
    int d  = i & (D_MODEL - 1);
    int bl = i >> 10;
    int l  = bl & (SEQ - 1);
    float v = pe[l * D_MODEL + d] + cond[i];
    x[i]  = v;
    xb[i] = f2bf(v);
}

__global__ void cvt_bf16_kernel(const float* __restrict__ src,
                                unsigned short* __restrict__ dst, int n)
{
    int i = blockIdx.x * blockDim.x + threadIdx.x;
    if (i < n) dst[i] = f2bf(src[i]);
}

// W_x [96, 2048] -> padded [128, 2048] bf16 with zero rows 96..127
__global__ void cvt_wx_pad_kernel(const float* __restrict__ src,
                                  unsigned short* __restrict__ dst)
{
    int i = blockIdx.x * blockDim.x + threadIdx.x;
    if (i >= NPAD_X * D_INNER) return;
    int r = i / D_INNER;
    dst[i] = (r < (DT_RANK + 2 * D_STATE)) ? f2bf(src[i]) : (unsigned short)0;
}

// dt = x_dbl[:, 0:64] (stride NPAD_X) -> bf16 [BL, 64]
__global__ void cvt_dt_kernel(const float* __restrict__ xdbl,
                              unsigned short* __restrict__ dtb)
{
    int i = blockIdx.x * blockDim.x + threadIdx.x;
    if (i >= BL * DT_RANK) return;
    int j  = i & (DT_RANK - 1);
    int bl = i >> 6;
    dtb[i] = f2bf(xdbl[(size_t)bl * NPAD_X + j]);
}

// causal depthwise conv (d_conv=4) over u = xz[:, :, 0:D_INNER], then SiLU
__global__ void conv_silu_kernel(const float* __restrict__ xz,
                                 const float* __restrict__ cw,
                                 const float* __restrict__ cb,
                                 float* __restrict__ xc,
                                 unsigned short* __restrict__ xcb)
{
    int i = blockIdx.x * blockDim.x + threadIdx.x;
    if (i >= BL * D_INNER) return;
    int c  = i & (D_INNER - 1);
    int bl = i >> 11;
    int l  = bl & (SEQ - 1);
    int b  = bl >> 10;
    float acc = cb[c];
    #pragma unroll
    for (int k = 0; k < D_CONV; ++k) {
        int lk = l + k - (D_CONV - 1);
        if (lk >= 0)
            acc += cw[c * D_CONV + k] *
                   xz[(size_t)(b * SEQ + lk) * (2 * D_INNER) + c];
    }
    float v = silu_f(acc);
    xc[i]  = v;
    xcb[i] = f2bf(v);
}

// delta = softplus(delta_raw + b_dt), in place
__global__ void softplus_bias_kernel(float* __restrict__ delta,
                                     const float* __restrict__ b_dt)
{
    int i = blockIdx.x * blockDim.x + threadIdx.x;
    if (i >= BL * D_INNER) return;
    int c = i & (D_INNER - 1);
    delta[i] = softplus_f(delta[i] + b_dt[c]);
}

// =====================================================================
// Chunked selective scan.
// Recurrence per (b, c, s):  h <- exp2(d*A2)*h + (d*u)*B[s],  y = sum_s h*C[s]
// with A2 = -exp(A_log)*log2(e) precomputed so the hot loop is bare v_exp_f32.
// Pass 1: per chunk with h0=0 -> (h_end, sum_d).   Chunk decay = exp2(A2*sum_d).
// Pass 2: 16-step prefix across chunks -> H0 entry states.
// Pass 3: re-scan each chunk seeded with H0, fused output gate -> bf16.
// The whole chunk's B (and C) block is staged in LDS ONCE (single barrier),
// so the 64-step sequential loop has no barriers; delta/xc/z are software-
// pipelined one step ahead in registers to hide global-load latency.
// =====================================================================
__global__ __launch_bounds__(256) void scan_pass1_kernel(
    const float* __restrict__ xc,     // [BL, D_INNER]
    const float* __restrict__ delta,  // [BL, D_INNER]
    const float* __restrict__ xdbl,   // [BL, NPAD_X], B at cols 64..79
    const float* __restrict__ A_log,  // [D_INNER, D_STATE]
    float* __restrict__ hend,         // [B, NCHUNK, D_INNER, D_STATE]
    float* __restrict__ sumd)         // [B, NCHUNK, D_INNER]
{
    const int CB = D_INNER / 256;
    const int cblk  = blockIdx.x % CB;
    const int chunk = (blockIdx.x / CB) % NCHUNK;
    const int b     = blockIdx.x / (CB * NCHUNK);
    const int c     = cblk * 256 + threadIdx.x;
    const size_t row0 = (size_t)(b * SEQ + chunk * CHUNK_L);

    // stage the whole chunk's B block: 64 steps x 16 states = 4 KB
    __shared__ float sB[CHUNK_L][D_STATE];
    for (int t = threadIdx.x; t < CHUNK_L * D_STATE; t += 256) {
        const int ll = t >> 4, ss = t & (D_STATE - 1);
        sB[ll][ss] = xdbl[(row0 + ll) * NPAD_X + DT_RANK + ss];
    }

    float A2[D_STATE];
    #pragma unroll
    for (int s = 0; s < D_STATE; ++s)
        A2[s] = -__expf(A_log[(size_t)c * D_STATE + s]) * LOG2E;

    float h[D_STATE];
    #pragma unroll
    for (int s = 0; s < D_STATE; ++s) h[s] = 0.0f;
    float sd = 0.0f;

    __syncthreads();   // single barrier; no barriers inside the scan loop

    float d_n = delta[row0 * D_INNER + c];
    float u_n = xc[row0 * D_INNER + c];
    for (int ll = 0; ll < CHUNK_L; ++ll) {
        const float d = d_n, u = u_n;
        if (ll + 1 < CHUNK_L) {                // prefetch next step
            d_n = delta[(row0 + ll + 1) * D_INNER + c];
            u_n = xc[(row0 + ll + 1) * D_INNER + c];
        }
        sd += d;
        const float du = d * u;
        #pragma unroll
        for (int s = 0; s < D_STATE; ++s)
            h[s] = exp2f(d * A2[s]) * h[s] + du * sB[ll][s];
    }

    const size_t base = (size_t)(b * NCHUNK + chunk) * D_INNER + c;
    #pragma unroll
    for (int s = 0; s < D_STATE; ++s)
        hend[base * D_STATE + s] = h[s];
    sumd[base] = sd;
}

__global__ void scan_pass2_kernel(
    const float* __restrict__ A_log,
    const float* __restrict__ hend,
    const float* __restrict__ sumd,
    float* __restrict__ H0)           // [B, NCHUNK, D_INNER, D_STATE]
{
    int i = blockIdx.x * blockDim.x + threadIdx.x;
    if (i >= BATCH * D_INNER * D_STATE) return;
    const int s = i & (D_STATE - 1);
    const int c = (i >> 4) & (D_INNER - 1);
    const int b = i >> 15;           // 4 + 11 bits

    const float A2 = -__expf(A_log[(size_t)c * D_STATE + s]) * LOG2E;
    float H = 0.0f;
    for (int j = 0; j < NCHUNK; ++j) {
        const size_t base = (size_t)(b * NCHUNK + j) * D_INNER + c;
        H0[base * D_STATE + s] = H;
        H = exp2f(A2 * sumd[base]) * H + hend[base * D_STATE + s];
    }
}

__global__ __launch_bounds__(256) void scan_pass3_kernel(
    const float* __restrict__ xc,
    const float* __restrict__ delta,
    const float* __restrict__ xdbl,   // B at 64..79, C at 80..95
    const float* __restrict__ xz,     // z = xz[:, :, D_INNER + c]
    const float* __restrict__ A_log,
    const float* __restrict__ Dp,
    const float* __restrict__ H0,
    unsigned short* __restrict__ yact) // [BL, D_INNER] bf16
{
    const int CB = D_INNER / 256;
    const int cblk  = blockIdx.x % CB;
    const int chunk = (blockIdx.x / CB) % NCHUNK;
    const int b     = blockIdx.x / (CB * NCHUNK);
    const int c     = cblk * 256 + threadIdx.x;
    const size_t row0 = (size_t)(b * SEQ + chunk * CHUNK_L);

    // stage the whole chunk's B and C blocks: 64 x 32 floats = 8 KB
    __shared__ float sB[CHUNK_L][D_STATE];
    __shared__ float sC[CHUNK_L][D_STATE];
    for (int t = threadIdx.x; t < CHUNK_L * 2 * D_STATE; t += 256) {
        const int ll = t >> 5;
        const int ss = t & (2 * D_STATE - 1);
        const float v = xdbl[(row0 + ll) * NPAD_X + DT_RANK + ss];
        if (ss < D_STATE) sB[ll][ss] = v;
        else              sC[ll][ss - D_STATE] = v;
    }

    float A2[D_STATE];
    #pragma unroll
    for (int s = 0; s < D_STATE; ++s)
        A2[s] = -__expf(A_log[(size_t)c * D_STATE + s]) * LOG2E;

    float h[D_STATE];
    const size_t base = (size_t)(b * NCHUNK + chunk) * D_INNER + c;
    #pragma unroll
    for (int s = 0; s < D_STATE; ++s)
        h[s] = H0[base * D_STATE + s];

    const float dcoef = Dp[c];

    __syncthreads();   // single barrier; scan loop below is barrier-free

    float d_n = delta[row0 * D_INNER + c];
    float u_n = xc[row0 * D_INNER + c];
    float z_n = xz[row0 * (2 * D_INNER) + D_INNER + c];
    for (int ll = 0; ll < CHUNK_L; ++ll) {
        const float d = d_n, u = u_n, z = z_n;
        if (ll + 1 < CHUNK_L) {                // prefetch next step
            d_n = delta[(row0 + ll + 1) * D_INNER + c];
            u_n = xc[(row0 + ll + 1) * D_INNER + c];
            z_n = xz[(row0 + ll + 1) * (2 * D_INNER) + D_INNER + c];
        }
        const float du = d * u;
        float y = 0.0f;
        #pragma unroll
        for (int s = 0; s < D_STATE; ++s) {
            h[s] = exp2f(d * A2[s]) * h[s] + du * sB[ll][s];
            y   += h[s] * sC[ll][s];
        }
        yact[(row0 + ll) * D_INNER + c] = f2bf((y + u * dcoef) * silu_f(z));
    }
}

// =====================================================================
// host-side orchestration
// =====================================================================
static inline int eb(int n) { return (n + 255) / 256; }

extern "C" void kernel_launch(void* const* d_in, const int* in_sizes, int n_in,
                              void* d_out, int out_size, void* d_ws, size_t ws_size,
                              hipStream_t stream)
{
    (void)in_sizes; (void)n_in; (void)out_size; (void)ws_size;
    const float* pe     = (const float*)d_in[0];
    const float* cond   = (const float*)d_in[1];
    const float* W_in   = (const float*)d_in[2];   // [L, 2*di, dm]
    const float* conv_w = (const float*)d_in[3];   // [L, di, 4]
    const float* conv_b = (const float*)d_in[4];   // [L, di]
    const float* W_x    = (const float*)d_in[5];   // [L, 96, di]
    const float* W_dt   = (const float*)d_in[6];   // [L, di, 64]
    const float* b_dt   = (const float*)d_in[7];   // [L, di]
    const float* A_log  = (const float*)d_in[8];   // [L, di, 16]
    const float* Dp     = (const float*)d_in[9];   // [L, di]
    const float* W_out  = (const float*)d_in[10];  // [L, dm, di]
    float* out          = (float*)d_out;           // [B, L, dm] f32

    // carve workspace
    char* ws = (char*)d_ws;
    auto carve = [&](size_t bytes) {
        char* p = ws;
        ws += (bytes + 255) & ~(size_t)255;
        return (void*)p;
    };
    float*          x_f32   = (float*)         carve((size_t)BL * D_MODEL * 4);
    unsigned short* x_bf    = (unsigned short*)carve((size_t)BL * D_MODEL * 2);
    float*          xz      = (float*)         carve((size_t)BL * 2 * D_INNER * 4);
    float*          xc      = (float*)         carve((size_t)BL * D_INNER * 4);
    unsigned short* xc_bf   = (unsigned short*)carve((size_t)BL * D_INNER * 2);
    float*          x_dbl   = (float*)         carve((size_t)BL * NPAD_X * 4);
    unsigned short* dt_bf   = (unsigned short*)carve((size_t)BL * DT_RANK * 2);
    float*          delta   = (float*)         carve((size_t)BL * D_INNER * 4);
    float*          hend    = (float*)         carve((size_t)BATCH * NCHUNK * D_INNER * D_STATE * 4);
    float*          H0      = (float*)         carve((size_t)BATCH * NCHUNK * D_INNER * D_STATE * 4);
    float*          sumd    = (float*)         carve((size_t)BATCH * NCHUNK * D_INNER * 4);
    unsigned short* yact_bf = (unsigned short*)carve((size_t)BL * D_INNER * 2);
    unsigned short* Win_bf  = (unsigned short*)carve((size_t)2 * D_INNER * D_MODEL * 2);
    unsigned short* Wx_bf   = (unsigned short*)carve((size_t)NPAD_X * D_INNER * 2);
    unsigned short* Wdt_bf  = (unsigned short*)carve((size_t)D_INNER * DT_RANK * 2);
    unsigned short* Wout_bf = (unsigned short*)carve((size_t)D_MODEL * D_INNER * 2);

    // x = pe + condition (f32 + bf16 copy)
    prep_x_kernel<<<eb(BL * D_MODEL), 256, 0, stream>>>(pe, cond, x_f32, x_bf);

    for (int layer = 0; layer < NUM_LAYERS; ++layer) {
        const float* Wi = W_in   + (size_t)layer * 2 * D_INNER * D_MODEL;
        const float* cw = conv_w + (size_t)layer * D_INNER * D_CONV;
        const float* cb = conv_b + (size_t)layer * D_INNER;
        const float* Wx = W_x    + (size_t)layer * (DT_RANK + 2 * D_STATE) * D_INNER;
        const float* Wd = W_dt   + (size_t)layer * D_INNER * DT_RANK;
        const float* bd = b_dt   + (size_t)layer * D_INNER;
        const float* Al = A_log  + (size_t)layer * D_INNER * D_STATE;
        const float* Dl = Dp     + (size_t)layer * D_INNER;
        const float* Wo = W_out  + (size_t)layer * D_MODEL * D_INNER;

        // weight conversions (bf16, row-major [N,K] as WMMA-B wants)
        cvt_bf16_kernel<<<eb(2 * D_INNER * D_MODEL), 256, 0, stream>>>(
            Wi, Win_bf, 2 * D_INNER * D_MODEL);
        cvt_wx_pad_kernel<<<eb(NPAD_X * D_INNER), 256, 0, stream>>>(Wx, Wx_bf);
        cvt_bf16_kernel<<<eb(D_INNER * DT_RANK), 256, 0, stream>>>(
            Wd, Wdt_bf, D_INNER * DT_RANK);
        cvt_bf16_kernel<<<eb(D_MODEL * D_INNER), 256, 0, stream>>>(
            Wo, Wout_bf, D_MODEL * D_INNER);

        // xz = x @ W_in^T : [2048,1024] x [4096,1024]^T
        {
            int tiles = (BL / 64) * ((2 * D_INNER) / 64);
            gemm_bf16_nt<<<(tiles + 7) / 8, 256, 0, stream>>>(
                x_bf, D_MODEL, Win_bf, D_MODEL, xz, 2 * D_INNER,
                BL, 2 * D_INNER, D_MODEL);
        }

        // xc = silu(causal_conv(u)), u = xz[:, :, :D_INNER]
        conv_silu_kernel<<<eb(BL * D_INNER), 256, 0, stream>>>(xz, cw, cb, xc, xc_bf);

        // x_dbl = xc @ W_x^T : [2048,2048] x [128(pad),2048]^T
        {
            int tiles = (BL / 64) * (NPAD_X / 64);
            gemm_bf16_nt<<<(tiles + 7) / 8, 256, 0, stream>>>(
                xc_bf, D_INNER, Wx_bf, D_INNER, x_dbl, NPAD_X,
                BL, NPAD_X, D_INNER);
        }

        // dt -> bf16
        cvt_dt_kernel<<<eb(BL * DT_RANK), 256, 0, stream>>>(x_dbl, dt_bf);

        // delta_raw = dt @ W_dt^T : [2048,64] x [2048,64]^T
        {
            int tiles = (BL / 64) * (D_INNER / 64);
            gemm_bf16_nt<<<(tiles + 7) / 8, 256, 0, stream>>>(
                dt_bf, DT_RANK, Wdt_bf, DT_RANK, delta, D_INNER,
                BL, D_INNER, DT_RANK);
        }
        softplus_bias_kernel<<<eb(BL * D_INNER), 256, 0, stream>>>(delta, bd);

        // chunked selective scan (depth SEQ/NCHUNK instead of SEQ)
        {
            const int blocks13 = BATCH * NCHUNK * (D_INNER / 256);
            scan_pass1_kernel<<<blocks13, 256, 0, stream>>>(
                xc, delta, x_dbl, Al, hend, sumd);
            scan_pass2_kernel<<<eb(BATCH * D_INNER * D_STATE), 256, 0, stream>>>(
                Al, hend, sumd, H0);
            scan_pass3_kernel<<<blocks13, 256, 0, stream>>>(
                xc, delta, x_dbl, xz, Al, Dl, H0, yact_bf);
        }

        // next_x = gated @ W_out^T : [2048,2048] x [1024,2048]^T
        float* gemm_dst = (layer == NUM_LAYERS - 1) ? out : x_f32;
        {
            int tiles = (BL / 64) * (D_MODEL / 64);
            gemm_bf16_nt<<<(tiles + 7) / 8, 256, 0, stream>>>(
                yact_bf, D_INNER, Wout_bf, D_INNER, gemm_dst, D_MODEL,
                BL, D_MODEL, D_INNER);
        }
        if (layer != NUM_LAYERS - 1) {
            cvt_bf16_kernel<<<eb(BL * D_MODEL), 256, 0, stream>>>(
                x_f32, x_bf, BL * D_MODEL);
        }
    }
}
